// RoPESelfAttentionBase_80470507257926
// MI455X (gfx1250) — matrix operand order repaired and checked by reading, not compiled, not run on hardware
//
#include <hip/hip_runtime.h>
#include <hip/hip_bf16.h>

// ---------------- problem constants ----------------
#define BB   4
#define TT   1024
#define EE   1024
#define HH   16
#define DD   64
#define BT   (BB * TT)          // 4096 rows
#define QK_SCALE 0.125f         // D^-0.5 = 1/8

typedef __attribute__((ext_vector_type(16))) _Float16 v16h;
typedef __attribute__((ext_vector_type(8)))  _Float16 v8h;
typedef __attribute__((ext_vector_type(8)))  float    v8f;

// ---------------- WMMA fragment helpers ----------------
// A fragment (16x32 f16, row-major source, leading dim lda):
//   lane 0-15 : M=lane,   K = {0..7} in v0..3, {16..23} in v4..7
//   lane 16-31: M=lane-16,K = {8..15} in v0..3, {24..31} in v4..7
static __device__ __forceinline__ v16h load_frag_a(const _Float16* base, int lda,
                                                   int row0, int k0) {
    int lane = threadIdx.x & 31;
    int m    = lane & 15;
    int kb   = (lane >> 4) << 3;          // 0 or 8
    const _Float16* p = base + (size_t)(row0 + m) * lda + k0 + kb;
    union { v16h v; v8h h[2]; } f;
    f.h[0] = *(const v8h*)p;              // K = kb .. kb+7
    f.h[1] = *(const v8h*)(p + 16);       // K = kb+16 .. kb+23
    return f.v;
}

// B fragment (32x16 f16, row-major source B[k][n], leading dim lda):
//   lane 0-15 : N=lane,   K = {0..7} in v0..3, {16..23} in v4..7
//   lane 16-31: N=lane-16,K = {8..15} in v0..3, {24..31} in v4..7
static __device__ __forceinline__ v16h load_frag_b(const _Float16* base, int lda,
                                                   int k0, int n0) {
    int lane = threadIdx.x & 31;
    int kb   = (lane >> 4) << 3;
    const _Float16* p = base + (size_t)(k0 + kb) * lda + n0 + (lane & 15);
    v16h v;
#pragma unroll
    for (int s = 0; s < 8; ++s) {
        v[s]     = p[(size_t)s * lda];
        v[8 + s] = p[(size_t)(16 + s) * lda];
    }
    return v;
}

static __device__ __forceinline__ int crow(int i) {      // C/D row for VGPR i
    return i + (((threadIdx.x & 31) >> 4) << 3);
}

static __device__ __forceinline__ v8f wmma_f16(v16h a, v16h b, v8f c) {
    return __builtin_amdgcn_wmma_f32_16x16x32_f16(false, a, false, b,
                                                  (short)0, c, false, false);
}

// ---------------- gfx1250 async global->LDS (ASYNCcnt) ----------------
static __device__ __forceinline__ void async_b128(const void* g, unsigned lds_byte_off) {
    asm volatile("global_load_async_to_lds_b128 %0, %1, off"
                 :: "v"(lds_byte_off), "v"(g) : "memory");
}
static __device__ __forceinline__ void wait_async0() {
    asm volatile("s_wait_asynccnt 0x0" ::: "memory");
}
static __device__ __forceinline__ void wait_async4() {
    asm volatile("s_wait_asynccnt 0x4" ::: "memory");
}
static __device__ __forceinline__ void wait_async8() {
    asm volatile("s_wait_asynccnt 0x8" ::: "memory");
}

// ---------------- kernel 1: f32 -> f16 pack of hidden_states ----------------
__global__ void pack_x_kernel(const float* __restrict__ x, _Float16* __restrict__ xh, int n) {
    for (int i = blockIdx.x * blockDim.x + threadIdx.x; i < n; i += gridDim.x * blockDim.x)
        xh[i] = (_Float16)x[i];
}

// ---------------- kernel 2: transpose + pack weights (Wt[k][n] = W[n][k]) ----
__global__ void pack_w_kernel(const float* __restrict__ Wq, const float* __restrict__ Wk,
                              const float* __restrict__ Wv, const float* __restrict__ Wo,
                              _Float16* __restrict__ WqT, _Float16* __restrict__ WkT,
                              _Float16* __restrict__ WvT, _Float16* __restrict__ WoT) {
    const int per = EE * EE;
    for (int i = blockIdx.x * blockDim.x + threadIdx.x; i < 4 * per;
         i += gridDim.x * blockDim.x) {
        int mat = i >> 20;                 // i / (1024*1024)
        int rem = i & (per - 1);
        int k   = rem >> 10;
        int n   = rem & 1023;
        const float* Ws = (mat == 0) ? Wq : (mat == 1) ? Wk : (mat == 2) ? Wv : Wo;
        _Float16*    Wd = (mat == 0) ? WqT : (mat == 1) ? WkT : (mat == 2) ? WvT : WoT;
        Wd[(size_t)k * EE + n] = (_Float16)Ws[(size_t)n * EE + k];
    }
}

// ---------------- kernel 3: fused QKV projection GEMM (WMMA) ----------------
// grid (64, 16, 3); 1 wave per 64x64 output tile; K = 1024 in 32 steps.
// 16 WMMA per 8 b128 A loads + 64 f16 B loads; launch_bounds(32,1) -> full VGPR file, no spills.
__global__ __launch_bounds__(32, 1)
void qkv_gemm_kernel(const _Float16* __restrict__ Xh,
                     const _Float16* __restrict__ WqT,
                     const _Float16* __restrict__ WkT,
                     const _Float16* __restrict__ WvT,
                     const float* __restrict__ bq,
                     const float* __restrict__ bk,
                     const float* __restrict__ bv,
                     float* __restrict__ q32, float* __restrict__ k32,
                     _Float16* __restrict__ vh) {
    const int m0   = blockIdx.x * 64;
    const int n0   = blockIdx.y * 64;
    const int proj = blockIdx.z;
    const _Float16* Wt = (proj == 0) ? WqT : (proj == 1) ? WkT : WvT;

    v8f acc[4][4] = {};
#pragma unroll 1
    for (int kt = 0; kt < EE; kt += 32) {
        v16h a[4];
#pragma unroll
        for (int mi = 0; mi < 4; ++mi) a[mi] = load_frag_a(Xh, EE, m0 + 16 * mi, kt);
#pragma unroll
        for (int ni = 0; ni < 4; ++ni) {
            v16h b = load_frag_b(Wt, EE, kt, n0 + 16 * ni);
#pragma unroll
            for (int mi = 0; mi < 4; ++mi)
                acc[mi][ni] = wmma_f16(a[mi], b, acc[mi][ni]);
        }
    }

    const int lane = threadIdx.x & 31;
#pragma unroll
    for (int mi = 0; mi < 4; ++mi)
#pragma unroll
        for (int ni = 0; ni < 4; ++ni)
#pragma unroll
            for (int i = 0; i < 8; ++i) {
                int m = m0 + 16 * mi + crow(i);
                int n = n0 + 16 * ni + (lane & 15);
                int b = m >> 10, t = m & 1023;        // m = b*T + t
                int h = n >> 6,  d = n & 63;          // n = h*D + d
                size_t idx = (((size_t)(b * HH + h) * TT) + t) * DD + d;
                float v = acc[mi][ni][i];
                if (proj == 0)      q32[idx] = (v + bq[n]) * QK_SCALE;
                else if (proj == 1) k32[idx] =  v + bk[n];
                else                vh[idx]  = (_Float16)(v + bv[n]);
            }
}

// ---------------- kernel 4: block-rotation via 4x4 expm -------------------
static __device__ __forceinline__ void mm4(float* Dst, const float* X, const float* Y) {
#pragma unroll
    for (int r = 0; r < 4; ++r)
#pragma unroll
        for (int c = 0; c < 4; ++c) {
            float s = 0.f;
#pragma unroll
            for (int k = 0; k < 4; ++k) s += X[r * 4 + k] * Y[k * 4 + c];
            Dst[r * 4 + c] = s;
        }
}

// grid = B*H*(T/256) = 256 blocks of 256 threads; A skew-symmetrized into LDS.
__global__ void rot_kernel(const float* __restrict__ positions,
                           const float* __restrict__ A_raw,
                           const float* __restrict__ q32,
                           const float* __restrict__ k32,
                           _Float16* __restrict__ qrh,     // [B,H,T,D]
                           _Float16* __restrict__ krTh) {  // [B,H,D,T]
    __shared__ float As[3 * 16 * 16];                      // [axis][nb][r*4+c]
    const int tid = threadIdx.x;
    const int bi  = blockIdx.x;
    const int tc  = bi & 3;
    const int h   = (bi >> 2) & 15;
    const int b   = bi >> 6;
    const int t   = tc * 256 + tid;

    for (int i = tid; i < 768; i += 256) {
        int a = i >> 8, rem = i & 255;
        int nb = rem >> 4, rc = rem & 15;
        int r = rc >> 2, c = rc & 3;
        size_t base = ((size_t)(h * 3 + a) * 16 + nb) * 16;
        As[i] = A_raw[base + r * 4 + c] - A_raw[base + c * 4 + r];
    }
    __syncthreads();

    float p0 = positions[(size_t)(b * TT + t) * 3 + 0];
    float p1 = positions[(size_t)(b * TT + t) * 3 + 1];
    float p2 = positions[(size_t)(b * TT + t) * 3 + 2];

    const size_t qbase = (((size_t)(b * HH + h) * TT) + t) * DD;
    const size_t kTbase = ((size_t)(b * HH + h) * DD) * TT;

    for (int nb = 0; nb < 16; ++nb) {
        // lnR block, pre-scaled by 1/16 (scaling & squaring, s = 4)
        float M[16];
#pragma unroll
        for (int j = 0; j < 16; ++j)
            M[j] = 0.0625f * (p0 * As[nb * 16 + j] +
                              p1 * As[256 + nb * 16 + j] +
                              p2 * As[512 + nb * 16 + j]);
        // 6-term Taylor of expm, then 4 squarings
        float R[16], Tm[16], Nw[16];
#pragma unroll
        for (int j = 0; j < 16; ++j) { R[j] = (j % 5 == 0) ? 1.f : 0.f; Tm[j] = R[j]; }
#pragma unroll
        for (int ord = 1; ord <= 6; ++ord) {
            mm4(Nw, Tm, M);
            float inv = 1.f / (float)ord;
#pragma unroll
            for (int j = 0; j < 16; ++j) { Tm[j] = Nw[j] * inv; R[j] += Tm[j]; }
        }
#pragma unroll
        for (int sq = 0; sq < 4; ++sq) {
            mm4(Nw, R, R);
#pragma unroll
            for (int j = 0; j < 16; ++j) R[j] = Nw[j];
        }
        // rotate q and k 4-blocks
        float qv[4], kv[4];
#pragma unroll
        for (int c = 0; c < 4; ++c) {
            qv[c] = q32[qbase + nb * 4 + c];
            kv[c] = k32[qbase + nb * 4 + c];
        }
#pragma unroll
        for (int r = 0; r < 4; ++r) {
            float qs = 0.f, ks = 0.f;
#pragma unroll
            for (int c = 0; c < 4; ++c) {
                qs += R[r * 4 + c] * qv[c];
                ks += R[r * 4 + c] * kv[c];
            }
            qrh[qbase + nb * 4 + r] = (_Float16)qs;
            krTh[kTbase + (size_t)(nb * 4 + r) * TT + t] = (_Float16)ks;  // transposed
        }
    }
}

// ---------------- kernel 5: attention -------------------------------------
// 4 waves/block; per-wave LDS: f16 score/prob row (32KB, exp overwrites scores
// in place), double-buffered async K-panel (2x2KB) and V-panel (2x4KB), scratch.
#define WPB      4
#define SP_OFF   0
#define KT_OFF   (16 * 1024 * 2)                 // 32768
#define V_OFF    (KT_OFF + 2 * 2048)             // 36864
#define SCR_OFF  (V_OFF + 2 * 4096)              // 45056
#define WAVE_LDS_BYTES (SCR_OFF + 512)           // 45568

__global__ __launch_bounds__(32 * WPB, 1)
void attn_kernel(const _Float16* __restrict__ qrh,
                 const _Float16* __restrict__ krTh,
                 const _Float16* __restrict__ vh,
                 _Float16* __restrict__ Oh) {   // [B,T,E] f16
    extern __shared__ char smem[];
    const int wave = threadIdx.x >> 5;
    const int lane = threadIdx.x & 31;

    char*     wb  = smem + (size_t)wave * WAVE_LDS_BYTES;
    _Float16* SPh = (_Float16*)(wb + SP_OFF);     // [16][1024] scores -> probs
    _Float16* KTb = (_Float16*)(wb + KT_OFF);     // [2][64][16]
    _Float16* Vb  = (_Float16*)(wb + V_OFF);      // [2][32][64]
    float*    scr = (float*)(wb + SCR_OFF);       // reduction scratch

    const int g  = blockIdx.x * WPB + wave;       // 0..4095
    const int bh = g >> 6;
    const int t0 = (g & 63) * 16;
    const int b  = bh >> 4;
    const int h  = bh & 15;

    const _Float16* Q  = qrh  + (size_t)bh * TT * DD;
    const _Float16* KT = krTh + (size_t)bh * DD * TT;
    const _Float16* V  = vh   + (size_t)bh * TT * DD;

    // async stage of K^T panel st (64 rows x 16 cols f16 = 2KB): 4 issues/wave
    auto issueKT = [&](int st, int buf) {
        const char* gb = (const char*)KT + (size_t)st * 32;   // col byte offset
#pragma unroll
        for (int j = 0; j < 4; ++j) {
            int c = j * 32 + lane;
            int row = c >> 1, half = c & 1;
            async_b128(gb + (size_t)row * (TT * 2) + half * 16,
                       (unsigned)(size_t)((char*)KTb + buf * 2048 + row * 32 + half * 16));
        }
    };
    // async stage of V panel st (32 rows x 64 cols f16 = 4KB, contiguous): 8 issues
    auto issueV = [&](int st, int buf) {
        const char* gb = (const char*)V + (size_t)st * 4096;
#pragma unroll
        for (int j = 0; j < 8; ++j) {
            int off = (j * 32 + lane) * 16;
            async_b128(gb + off, (unsigned)(size_t)((char*)Vb + buf * 4096 + off));
        }
    };

    // A fragments of the 16-row query tile (K = 64 -> two fragments), reused.
    v16h a0 = load_frag_a(Q, DD, t0, 0);
    v16h a1 = load_frag_a(Q, DD, t0, 32);

    // ---- phase 1: all scores for this tile into LDS (f16) ----
    issueKT(0, 0);
    for (int st = 0; st < 64; ++st) {
        int buf = st & 1;
        if (st < 63) { issueKT(st + 1, buf ^ 1); wait_async4(); }
        else         { wait_async0(); }
        v16h b0 = load_frag_b(KTb + buf * 1024, 16, 0,  0);
        v16h b1 = load_frag_b(KTb + buf * 1024, 16, 32, 0);
        v8f z = {};
        v8f s = wmma_f16(a0, b0, z);
        s     = wmma_f16(a1, b1, s);
#pragma unroll
        for (int i = 0; i < 8; ++i)
            SPh[(size_t)crow(i) * 1024 + st * 16 + (lane & 15)] = (_Float16)s[i];
    }
    __syncthreads();

    // ---- phase 2: softmax over full 1024-wide rows (f32 math, f16 storage) ----
    const int r  = lane & 15;
    const int hh = lane >> 4;
    const int c0 = hh * 512;

    float lmax = -3.0e38f;
    for (int c = 0; c < 512; ++c)
        lmax = fmaxf(lmax, (float)SPh[(size_t)r * 1024 + c0 + c]);
    scr[hh * 16 + r] = lmax;
    __syncthreads();
    float rmax = fmaxf(scr[r], scr[16 + r]);

    float lsum = 0.f;
    for (int c = 0; c < 512; ++c) {
        float e = __expf((float)SPh[(size_t)r * 1024 + c0 + c] - rmax);
        lsum += e;
        SPh[(size_t)r * 1024 + c0 + c] = (_Float16)e;   // unnormalized, in place
    }
    scr[32 + hh * 16 + r] = lsum;
    __syncthreads();
    float rsum = scr[32 + r] + scr[48 + r];
    if (hh == 0) scr[r] = 1.f / rsum;                   // per-row inverse sum
    __syncthreads();

    // ---- phase 3: O = P @ V (normalization deferred to store) ----
    v8f o[4] = {};
    issueV(0, 0);
    for (int st = 0; st < 32; ++st) {
        int buf = st & 1;
        if (st < 31) { issueV(st + 1, buf ^ 1); wait_async8(); }
        else         { wait_async0(); }
        v16h pa = load_frag_a(SPh, 1024, 0, st * 32);   // ds_load_b128 x2
#pragma unroll
        for (int dt = 0; dt < 4; ++dt) {
            v16h vb = load_frag_b(Vb + buf * 2048, DD, 0, dt * 16);
            o[dt] = wmma_f16(pa, vb, o[dt]);
        }
    }

#pragma unroll
    for (int dt = 0; dt < 4; ++dt)
#pragma unroll
        for (int i = 0; i < 8; ++i) {
            int m   = crow(i);
            int row = t0 + m;
            int col = dt * 16 + (lane & 15);
            Oh[((size_t)(b * TT + row)) * EE + h * DD + col] =
                (_Float16)(o[dt][i] * scr[m]);
        }
}

// ---------------- kernel 6: output projection GEMM (64x64/wave) -------------
__global__ __launch_bounds__(32, 1)
void out_gemm_kernel(const _Float16* __restrict__ Oh,
                     const _Float16* __restrict__ WoT,
                     const float* __restrict__ bo,
                     float* __restrict__ out) {
    const int m0 = blockIdx.x * 64;
    const int n0 = blockIdx.y * 64;

    v8f acc[4][4] = {};
#pragma unroll 1
    for (int kt = 0; kt < EE; kt += 32) {
        v16h a[4];
#pragma unroll
        for (int mi = 0; mi < 4; ++mi) a[mi] = load_frag_a(Oh, EE, m0 + 16 * mi, kt);
#pragma unroll
        for (int ni = 0; ni < 4; ++ni) {
            v16h b = load_frag_b(WoT, EE, kt, n0 + 16 * ni);
#pragma unroll
            for (int mi = 0; mi < 4; ++mi)
                acc[mi][ni] = wmma_f16(a[mi], b, acc[mi][ni]);
        }
    }
    const int lane = threadIdx.x & 31;
#pragma unroll
    for (int mi = 0; mi < 4; ++mi)
#pragma unroll
        for (int ni = 0; ni < 4; ++ni)
#pragma unroll
            for (int i = 0; i < 8; ++i) {
                int m = m0 + 16 * mi + crow(i);
                int n = n0 + 16 * ni + (lane & 15);
                out[(size_t)m * EE + n] = acc[mi][ni][i] + bo[n];
            }
}

// ---------------- host launch ----------------
extern "C" void kernel_launch(void* const* d_in, const int* in_sizes, int n_in,
                              void* d_out, int out_size, void* d_ws, size_t ws_size,
                              hipStream_t stream) {
    const float* hs   = (const float*)d_in[0];
    const float* pos  = (const float*)d_in[1];
    const float* Araw = (const float*)d_in[2];
    const float* Wq   = (const float*)d_in[3];
    const float* bq   = (const float*)d_in[4];
    const float* Wk   = (const float*)d_in[5];
    const float* bk   = (const float*)d_in[6];
    const float* Wv   = (const float*)d_in[7];
    const float* bv   = (const float*)d_in[8];
    const float* Wo   = (const float*)d_in[9];
    const float* bo   = (const float*)d_in[10];
    float* out = (float*)d_out;

    // workspace carve-up (all 256B aligned)
    char* ws = (char*)d_ws;
    size_t off = 0;
    auto carve = [&](size_t bytes) { char* p = ws + off; off += (bytes + 255) & ~(size_t)255; return p; };
    _Float16* Xh   = (_Float16*)carve((size_t)BT * EE * 2);
    _Float16* WqT  = (_Float16*)carve((size_t)EE * EE * 2);
    _Float16* WkT  = (_Float16*)carve((size_t)EE * EE * 2);
    _Float16* WvT  = (_Float16*)carve((size_t)EE * EE * 2);
    _Float16* WoT  = (_Float16*)carve((size_t)EE * EE * 2);
    float*    q32  = (float*)   carve((size_t)BT * EE * 4);
    float*    k32  = (float*)   carve((size_t)BT * EE * 4);
    _Float16* vh   = (_Float16*)carve((size_t)BT * EE * 2);
    _Float16* qrh  = (_Float16*)carve((size_t)BT * EE * 2);
    _Float16* krTh = (_Float16*)carve((size_t)BT * EE * 2);
    _Float16* Oh   = (_Float16*)carve((size_t)BT * EE * 2);
    (void)ws_size; (void)in_sizes; (void)n_in; (void)out_size;

    // 1) pack activations and weights to f16 (weights transposed)
    pack_x_kernel<<<2048, 256, 0, stream>>>(hs, Xh, BT * EE);
    pack_w_kernel<<<2048, 256, 0, stream>>>(Wq, Wk, Wv, Wo, WqT, WkT, WvT, WoT);

    // 2) fused QKV projections (64x64 tile per wave)
    qkv_gemm_kernel<<<dim3(BT / 64, EE / 64, 3), 32, 0, stream>>>(
        Xh, WqT, WkT, WvT, bq, bk, bv, q32, k32, vh);

    // 3) per-token block rotations (expm of 4x4 skew blocks)
    rot_kernel<<<BB * HH * (TT / 256), 256, 0, stream>>>(pos, Araw, q32, k32, qrh, krTh);

    // 4) attention: 4 waves/block, async double-buffered K/V panels
    attn_kernel<<<(BB * HH * (TT / 16)) / WPB, 32 * WPB, WPB * WAVE_LDS_BYTES, stream>>>(
        qrh, krTh, vh, Oh);

    // 5) output projection (64x64 tile per wave)
    out_gemm_kernel<<<dim3(BT / 64, EE / 64), 32, 0, stream>>>(Oh, WoT, bo, out);
}
